// FullAttention_5901285065277
// MI455X (gfx1250) — matrix operand order repaired and checked by reading, not compiled, not run on hardware
//
#include <hip/hip_runtime.h>

typedef __attribute__((ext_vector_type(16))) __bf16 v16bf;
typedef __attribute__((ext_vector_type(8)))  __bf16 v8bf;
typedef __attribute__((ext_vector_type(4)))  __bf16 v4bf;
typedef __attribute__((ext_vector_type(2)))  __bf16 v2bf;
typedef __attribute__((ext_vector_type(8)))  float  v8f;
typedef __attribute__((ext_vector_type(4)))  float  f32x4;
typedef __attribute__((ext_vector_type(8)))  unsigned int u32x8;
typedef unsigned int u32;

namespace {
constexpr int Bc = 4, Lc = 2048, Hc = 16, Ec = 64;
constexpr int QTILE = 64;   // query rows per block (4 waves x 16)
constexpr int KTILE = 32;   // keys per iteration
constexpr int KRS   = 72;   // K LDS row stride (bf16 elems), 64 + 8 pad
constexpr int VRS   = 40;   // V^T LDS row stride (bf16 elems), 32 + 8 pad
// scale = 1/sqrt(E) folded with log2(e) so softmax runs in exp2 domain
constexpr float QSCALE = 0.125f * 1.44269504088896f;
}

__global__ __launch_bounds__(128)
void fa_fwd_causal(const float* __restrict__ Q,
                   const float* __restrict__ K,
                   const float* __restrict__ V,
                   float* __restrict__ O)
{
  // double-buffered tiles: K [key][kd], V^T [d][k]
  __shared__ __align__(16) __bf16 ks[2][KTILE * KRS];  // 2 x 4608 B
  __shared__ __align__(16) __bf16 vs[2][Ec * VRS];     // 2 x 5120 B

  const int tid   = threadIdx.x;
  // force wave index into an SGPR so uniform guards compile to SALU branches
  const int wave  = __builtin_amdgcn_readfirstlane(tid >> 5);
  const int lane  = tid & 31;
  const int l16   = lane & 15;
  const int lhalf = lane >> 4;

  const int nqb = Lc / QTILE;
  const int qb  = blockIdx.x % nqb;
  const int bh  = blockIdx.x / nqb;
  const int h   = bh % Hc;
  const int b   = bh / Hc;

  const int qbase_blk = qb * QTILE;
  const int qbase     = qbase_blk + wave * 16;  // SGPR (wave-uniform)
  const int q_row     = qbase + l16;            // this lane's query (N index)
  const int q_end     = qbase + 15;             // SGPR

  // [B,L,H,E]: element (b,l,h,d) at b*L*H*E + l*H*E + h*E + d
  const size_t base0   = (size_t)b * Lc * Hc * Ec + (size_t)h * Ec;
  const size_t lstride = (size_t)Hc * Ec;

  // ---- Q^T B-fragments (pre-scaled into exp2 domain) ----
  // B layout: lane = q + 16*(kd/16), elem e = kd%16
  v16bf bq[2];
  {
    const float* qrow = Q + base0 + (size_t)q_row * lstride;
    #pragma unroll
    for (int c = 0; c < 2; ++c) {
      const float* p = qrow + 32 * c + 16 * lhalf;
      #pragma unroll
      for (int j = 0; j < 4; ++j) {
        f32x4 x = *(const f32x4*)(p + 4 * j);
        #pragma unroll
        for (int i = 0; i < 4; ++i)
          bq[c][4 * j + i] = (__bf16)(x[i] * QSCALE);
      }
    }
  }

  // staging thread maps (block-cooperative)
  const int sk_key = tid >> 2;          // 0..31 : K row
  const int sk_dg  = (tid & 3) * 16;    // K dim group base
  const int sv_kq  = tid >> 4;          // 0..7  : V key quad base/4
  const int sv_dg  = (tid & 15) * 4;    // V dim quad base

  f32x4 kreg[4], vreg[4];               // prefetch registers

  auto load_tile = [&](int kb) {
    const float* krow = K + base0 + (size_t)(kb + sk_key) * lstride + sk_dg;
    #pragma unroll
    for (int j = 0; j < 4; ++j) kreg[j] = *(const f32x4*)(krow + 4 * j);
    #pragma unroll
    for (int j = 0; j < 4; ++j)
      vreg[j] = *(const f32x4*)(V + base0 + (size_t)(kb + 4 * sv_kq + j) * lstride + sv_dg);
  };

  auto store_tile = [&](int buf) {
    // K: [key][kd] contiguous bf16, two b128 stores
    #pragma unroll
    for (int jj = 0; jj < 2; ++jj) {
      v8bf w;
      #pragma unroll
      for (int i = 0; i < 8; ++i) w[i] = (__bf16)kreg[2 * jj + (i >> 2)][i & 3];
      *(v8bf*)(ks[buf] + sk_key * KRS + sk_dg + 8 * jj) = w;
    }
    // V^T: for each of 4 dims, 4 keys packed -> b64 store
    #pragma unroll
    for (int i = 0; i < 4; ++i) {
      v4bf w;
      #pragma unroll
      for (int j = 0; j < 4; ++j) w[j] = (__bf16)vreg[j][i];
      *(v4bf*)(vs[buf] + (sv_dg + i) * VRS + 4 * sv_kq) = w;
    }
  };

  v8f acc[4];                       // O^T tiles over d: [16t..16t+16) x 16 q
  #pragma unroll
  for (int t = 0; t < 4; ++t)
    #pragma unroll
    for (int r = 0; r < 8; ++r) acc[t][r] = 0.0f;

  float m_run = -__builtin_inff();
  float l_half = 0.0f;              // this half-wave's partial row sum

  const int kb_end = qbase_blk + QTILE;

  // prologue: stage tile 0 into buffer 0
  load_tile(0);
  store_tile(0);
  __syncthreads();

  for (int kb = 0; kb < kb_end; kb += KTILE) {
    const int  cur       = (kb >> 5) & 1;
    const bool have_next = (kb + KTILE < kb_end);
    if (have_next) load_tile(kb + KTILE);   // global loads fly during compute

    if (kb <= q_end) {   // scalar guard (q_end in SGPR): EXEC untouched
      // ---- fetch ALL K fragments first (8 x ds_load_b128), then 4 WMMAs
      // A layout: lane M = l16 (key), kd = 16*lhalf + e
      v16bf ak[2][2];
      #pragma unroll
      for (int kt = 0; kt < 2; ++kt) {
        const __bf16* krow = ks[cur] + (16 * kt + l16) * KRS + 16 * lhalf;
        #pragma unroll
        for (int cc = 0; cc < 2; ++cc) {
          v8bf lo = *(const v8bf*)(krow + 32 * cc);
          v8bf hi = *(const v8bf*)(krow + 32 * cc + 8);
          ak[kt][cc] = __builtin_shufflevector(lo, hi,
                         0, 1, 2, 3, 4, 5, 6, 7, 8, 9, 10, 11, 12, 13, 14, 15);
        }
      }
      // ---- S^T = K x Q^T : two 16(keys)x16(q) f32 tiles, K-dim = E in 2 chunks
      v8f st[2];
      #pragma unroll
      for (int kt = 0; kt < 2; ++kt) {
        v8f c;
        #pragma unroll
        for (int r = 0; r < 8; ++r) c[r] = 0.0f;
        c = __builtin_amdgcn_wmma_f32_16x16x32_bf16(false, ak[kt][0], false, bq[0],
                                                    (short)0, c, false, false);
        c = __builtin_amdgcn_wmma_f32_16x16x32_bf16(false, ak[kt][1], false, bq[1],
                                                    (short)0, c, false, false);
        st[kt] = c;
      }

      // ---- causal mask (key <= query), C layout: lane q=l16, k = 16*kt+8*lhalf+r
      if (kb + KTILE - 1 > qbase) {   // scalar condition
        #pragma unroll
        for (int kt = 0; kt < 2; ++kt)
          #pragma unroll
          for (int r = 0; r < 8; ++r) {
            const int key = kb + 16 * kt + 8 * lhalf + r;
            if (key > q_row) st[kt][r] = -__builtin_inff();
          }
      }

      // ---- online softmax in exp2 domain (per q = l16)
      float mt = m_run;
      #pragma unroll
      for (int kt = 0; kt < 2; ++kt)
        #pragma unroll
        for (int r = 0; r < 8; ++r) mt = fmaxf(mt, st[kt][r]);
      mt = fmaxf(mt, __shfl_xor(mt, 16, 32));   // both halves share mt

      const float alpha = exp2f(m_run - mt);    // identical in both halves
      float p0[8], p1[8];
      float rsum = 0.0f;
      #pragma unroll
      for (int r = 0; r < 8; ++r) {
        p0[r] = exp2f(st[0][r] - mt);
        p1[r] = exp2f(st[1][r] - mt);
        rsum += p0[r] + p1[r];
      }
      l_half = l_half * alpha + rsum;           // half-combine deferred to epilogue
      m_run = mt;
      #pragma unroll
      for (int t = 0; t < 4; ++t)
        #pragma unroll
        for (int r = 0; r < 8; ++r) acc[t][r] *= alpha;

      // ---- build P^T B operand: pack bf16 pairs FIRST, then one half-swap
      // B lane<16: q=lane, k=e (ST0); lane>=16: q=lane-16, k=16+e (ST1)
      u32 pa[4], pb[4];
      #pragma unroll
      for (int i = 0; i < 4; ++i) {
        v2bf t0; t0[0] = (__bf16)p0[2 * i]; t0[1] = (__bf16)p0[2 * i + 1];
        v2bf t1; t1[0] = (__bf16)p1[2 * i]; t1[1] = (__bf16)p1[2 * i + 1];
        pa[i] = __builtin_bit_cast(u32, t0);
        pb[i] = __builtin_bit_cast(u32, t1);
      }
      u32x8 wv;
      #pragma unroll
      for (int i = 0; i < 4; ++i) {
        const u32 ax = (u32)__shfl_xor((int)pa[i], 16, 32);
        const u32 bx = (u32)__shfl_xor((int)pb[i], 16, 32);
        wv[i]     = lhalf ? bx    : pa[i];
        wv[4 + i] = lhalf ? pb[i] : ax;
      }
      v16bf bp = __builtin_bit_cast(v16bf, wv);

      // ---- fetch ALL V^T fragments (8 x ds_load_b128), then 4 back-to-back WMMAs
      v16bf av[4];
      #pragma unroll
      for (int t = 0; t < 4; ++t) {
        const __bf16* vp = vs[cur] + (16 * t + l16) * VRS + 16 * lhalf;
        v8bf lo = *(const v8bf*)(vp);        // k = 16*lhalf .. +7
        v8bf hi = *(const v8bf*)(vp + 8);    // k = 16*lhalf+8 .. +7
        av[t] = __builtin_shufflevector(lo, hi,
                  0, 1, 2, 3, 4, 5, 6, 7, 8, 9, 10, 11, 12, 13, 14, 15);
      }
      #pragma unroll
      for (int t = 0; t < 4; ++t)
        acc[t] = __builtin_amdgcn_wmma_f32_16x16x32_bf16(false, av[t], false, bp,
                                                         (short)0, acc[t], false, false);
    }

    if (have_next) store_tile(cur ^ 1);   // other buffer: no race with readers
    __syncthreads();                      // single barrier per iteration
  }

  // ---- epilogue: combine half-sums, normalize, store O[b][q][h][d]
  const float l_tot = l_half + __shfl_xor(l_half, 16, 32);
  const float inv_l = 1.0f / l_tot;
  float* orow = O + base0 + (size_t)q_row * lstride;
  #pragma unroll
  for (int t = 0; t < 4; ++t) {
    f32x4 o0, o1;
    #pragma unroll
    for (int i = 0; i < 4; ++i) {
      o0[i] = acc[t][i]     * inv_l;
      o1[i] = acc[t][4 + i] * inv_l;
    }
    float* p = orow + 16 * t + 8 * lhalf;   // d = 16t + 8*lhalf + r
    *(f32x4*)(p)     = o0;
    *(f32x4*)(p + 4) = o1;
  }
}

extern "C" void kernel_launch(void* const* d_in, const int* in_sizes, int n_in,
                              void* d_out, int out_size, void* d_ws, size_t ws_size,
                              hipStream_t stream) {
  const float* Q = (const float*)d_in[0];
  const float* K = (const float*)d_in[1];
  const float* V = (const float*)d_in[2];
  // d_in[3] = attn_mask (bool causal) -- computed analytically in-kernel.
  float* O = (float*)d_out;

  const int blocks = Bc * Hc * (Lc / QTILE);  // 4*16*32 = 2048
  fa_fwd_causal<<<dim3(blocks), dim3(128), 0, stream>>>(Q, K, V, O);
}
